// EntityAwareSelfAttention_84670985273873
// MI455X (gfx1250) — matrix operand order repaired and checked by reading, not compiled, not run on hardware
//
#include <hip/hip_runtime.h>

// ---------------------------------------------------------------------------
// EntityAwareSelfAttention (LUKE-style) for gfx1250 / MI455X.
// fp32 in -> bf16 WMMA (v_wmma_f32_16x16x32_bf16, f32 accum) -> fp32 out.
// Uses gfx1250 async global->LDS copies (ASYNCcnt) when the builtin exists.
// ---------------------------------------------------------------------------

#define HID   1024
#define NHEAD 16
#define DH    64
#define SW    2048
#define SE    256
#define STOT  2304   // SW + SE
#define NBAT  2

typedef __attribute__((ext_vector_type(16))) __bf16 v16bf;
typedef __attribute__((ext_vector_type(8)))  float  v8f;

union Frag16 { v16bf v; unsigned u[8]; };

// ------------------------ async global->LDS copy ---------------------------
#if defined(__has_builtin)
#  if __has_builtin(__builtin_amdgcn_global_load_async_to_lds_b128)
#    define HAVE_ASYNC_LDS 1
#  endif
#endif
#ifndef HAVE_ASYNC_LDS
#  define HAVE_ASYNC_LDS 0
#endif

typedef int v4i_vs __attribute__((vector_size(16)));   // matches builtin proto

__device__ __forceinline__ void cp_b128(const void* g, void* l) {
#if HAVE_ASYNC_LDS
  __builtin_amdgcn_global_load_async_to_lds_b128(
      (__attribute__((address_space(1))) v4i_vs*)(v4i_vs*)(void*)g,
      (__attribute__((address_space(3))) v4i_vs*)(v4i_vs*)l,
      0, 0);
#else
  *(uint4*)l = *(const uint4*)g;
#endif
}

__device__ __forceinline__ void cp_wait() {
#if HAVE_ASYNC_LDS
#  if __has_builtin(__builtin_amdgcn_s_wait_asynccnt)
  __builtin_amdgcn_s_wait_asynccnt(0);
#  else
  asm volatile("s_wait_asynccnt 0x0" ::: "memory");
#  endif
#endif
}

__device__ __forceinline__ __bf16 f2bf(float f) {
  union { float f; unsigned u; } a; a.f = f;
  unsigned r = a.u + 0x7fffu + ((a.u >> 16) & 1u);   // round-to-nearest-even
  unsigned short h = (unsigned short)(r >> 16);
  union { unsigned short s; __bf16 b; } c; c.s = h;
  return c.b;
}

__device__ __forceinline__ v8f wmma_bf16(v16bf a, v16bf b, v8f c) {
  return __builtin_amdgcn_wmma_f32_16x16x32_bf16(false, a, false, b,
                                                 (short)0, c, false, false);
}

// A-fragment (16x32 bf16, MxK). Lane l: row = l&15; lanes 0-15 K-half 0,
// lanes 16-31 K-half 8. VGPR v<4 -> K = kh+2v ; v>=4 -> K = 16+kh+2(v-4).
__device__ __forceinline__ v16bf fragA_ld(const __bf16* base, size_t ld, int lane) {
  Frag16 f;
  int r  = lane & 15;
  int kh = (lane >> 4) * 8;
  const __bf16* p = base + (size_t)r * ld;
#pragma unroll
  for (int v = 0; v < 8; ++v) {
    int kp = (v < 4) ? (kh + 2 * v) : (16 + kh + 2 * (v - 4));
    f.u[v] = *(const unsigned*)(p + kp);
  }
  return f.v;
}

// B-fragment (32x16 bf16, KxN) read from [col][contraction] storage.
__device__ __forceinline__ v16bf fragB_ld(const __bf16* base, size_t ld, int lane) {
  Frag16 f;
  int n  = lane & 15;
  int kh = (lane >> 4) * 8;
  const __bf16* p = base + (size_t)n * ld;
#pragma unroll
  for (int v = 0; v < 8; ++v) f.u[v] = *(const unsigned*)(p + ((v < 4) ? (kh + 2 * v) : (16 + kh + 2 * (v - 4))));
  return f.v;
}

// ------------------------- conversion kernels ------------------------------
__global__ __launch_bounds__(256) void cvt_act(const float* __restrict__ in,
                                               __bf16* __restrict__ out,
                                               int perB, long inStrideB,
                                               long outStrideB, int total) {
  int idx = blockIdx.x * 256 + threadIdx.x;
  if (idx >= total) return;
  int  b = idx / perB;
  int  r = idx - b * perB;
  out[(long)b * outStrideB + r] = f2bf(in[(long)b * inStrideB + r]);
}

// Weights fp32 [k][n] -> bf16 transposed Wt[n][k]
__global__ __launch_bounds__(256) void cvt_w_t(const float* __restrict__ in,
                                               __bf16* __restrict__ out) {
  int idx = blockIdx.x * 256 + threadIdx.x;   // grid covers HID*HID exactly
  int k = idx >> 10, n = idx & 1023;
  out[(size_t)n * HID + k] = f2bf(in[idx]);
}

// ------------------------------ GEMM ---------------------------------------
// Y[M,1024] = X[M,1024] @ W + bias; W given transposed (Wt[n][k], bf16).
// Block: 256 thr (8 waves), tile 128x128; wave tile 64x32 = 4x2 WMMA tiles.
__device__ __forceinline__ void gemm_tile_cp(const __bf16* __restrict__ X,
                                             const __bf16* __restrict__ Wt,
                                             int m0, int n0, int kt,
                                             __bf16 (*As)[48], __bf16 (*Bs)[48],
                                             int tid) {
#pragma unroll
  for (int i = 0; i < 2; ++i) {
    int c = tid + 256 * i, row = c >> 2, off = (c & 3) * 8;
    cp_b128(X  + (size_t)(m0 + row) * HID + kt * 32 + off, &As[row][off]);
    cp_b128(Wt + (size_t)(n0 + row) * HID + kt * 32 + off, &Bs[row][off]);
  }
}

__global__ __launch_bounds__(256) void gemm_bf16(const __bf16* __restrict__ X,
                                                 const __bf16* __restrict__ Wt,
                                                 const float*  __restrict__ bias,
                                                 __bf16* __restrict__ Y,
                                                 int ldo, int transOut) {
  __shared__ __bf16 As[2][128][48];
  __shared__ __bf16 Bs[2][128][48];

  const int tid  = threadIdx.x;
  const int lane = tid & 31;
  const int w    = tid >> 5;
  const int wm   = w >> 2;          // 0..1
  const int wn   = w & 3;           // 0..3
  const int m0   = blockIdx.y * 128;
  const int n0   = blockIdx.x * 128;

  v8f acc[4][2];
  const v8f vz = {0.f,0.f,0.f,0.f,0.f,0.f,0.f,0.f};
#pragma unroll
  for (int i = 0; i < 4; ++i)
#pragma unroll
    for (int j = 0; j < 2; ++j) acc[i][j] = vz;

  const int nk = HID / 32;          // 32 K-steps

  gemm_tile_cp(X, Wt, m0, n0, 0, As[0], Bs[0], tid);
  cp_wait();
  __syncthreads();

  for (int kt = 0; kt < nk; ++kt) {
    const int buf = kt & 1;
    if (kt + 1 < nk)
      gemm_tile_cp(X, Wt, m0, n0, kt + 1, As[buf ^ 1], Bs[buf ^ 1], tid);
    if (kt + 2 < nk) {   // gfx1250 global_prefetch_b8 path
      int row = tid >> 2, off = (tid & 3) * 8;
      __builtin_prefetch(X  + (size_t)(m0 + row) * HID + (kt + 2) * 32 + off, 0, 3);
      __builtin_prefetch(Wt + (size_t)(n0 + row) * HID + (kt + 2) * 32 + off, 0, 3);
    }

    v16bf afr[4], bfr[2];
#pragma unroll
    for (int mt = 0; mt < 4; ++mt)
      afr[mt] = fragA_ld(&As[buf][wm * 64 + mt * 16][0], 48, lane);
#pragma unroll
    for (int nt = 0; nt < 2; ++nt)
      bfr[nt] = fragB_ld(&Bs[buf][wn * 32 + nt * 16][0], 48, lane);

#pragma unroll
    for (int mt = 0; mt < 4; ++mt)
#pragma unroll
      for (int nt = 0; nt < 2; ++nt)
        acc[mt][nt] = wmma_bf16(afr[mt], bfr[nt], acc[mt][nt]);

    cp_wait();
    __syncthreads();
  }

  // epilogue: C/D layout -> lane l holds col n = l&15, rows r + 8*(l>>4)
#pragma unroll
  for (int mt = 0; mt < 4; ++mt) {
#pragma unroll
    for (int nt = 0; nt < 2; ++nt) {
      int ncol = n0 + wn * 32 + nt * 16 + (lane & 15);
      float bc = bias[ncol];
#pragma unroll
      for (int r = 0; r < 8; ++r) {
        int row = m0 + wm * 64 + mt * 16 + r + ((lane >> 4) << 3);
        float val = acc[mt][nt][r] + bc;
        if (!transOut) Y[(size_t)row  * ldo + ncol] = f2bf(val);
        else           Y[(size_t)ncol * ldo + row ] = f2bf(val);
      }
    }
  }
}

// --------------------------- flash attention -------------------------------
// grid.x = NBAT*NHEAD*(STOT/64); block = 128 (4 waves); each wave: 16 queries.
// K/V tiles staged once per block into LDS (double buffered, async path).
__device__ __forceinline__ void kv_tile_cp(const __bf16* __restrict__ kBase,
                                           const __bf16* __restrict__ vBase,
                                           int s0, __bf16 (*Ks)[64],
                                           __bf16 (*Vs)[32], int tid) {
#pragma unroll
  for (int i = 0; i < 2; ++i) {           // K tile: 32 keys x 64 d
    int c = tid + 128 * i;
    int row = c >> 3, off = (c & 7) * 8;
    cp_b128(kBase + (size_t)(s0 + row) * HID + off, &Ks[row][off]);
  }
#pragma unroll
  for (int i = 0; i < 2; ++i) {           // V tile: 64 d x 32 keys (from Vt)
    int c = tid + 128 * i;
    int d = c >> 2, soff = (c & 3) * 8;
    cp_b128(vBase + (size_t)d * STOT + s0 + soff, &Vs[d][soff]);
  }
}

__global__ __launch_bounds__(128) void attn_fwd(const __bf16* __restrict__ Qa,
                                                const __bf16* __restrict__ Qb,
                                                const __bf16* __restrict__ Kk,
                                                const __bf16* __restrict__ Vt,
                                                float* __restrict__ outW,
                                                float* __restrict__ outE) {
  __shared__ __bf16 Ks[2][32][64];
  __shared__ __bf16 Vs[2][64][32];
  __shared__ __bf16 Plds[4][16][32];

  const int tid  = threadIdx.x;
  const int lane = tid & 31;
  const int w    = tid >> 5;
  int bid = blockIdx.x;
  const int nqt = STOT / 64;
  const int qt  = bid % nqt;  bid /= nqt;
  const int h   = bid % NHEAD;
  const int b   = bid / NHEAD;
  const int q0  = qt * 64 + w * 16;

  const __bf16* qaBase = Qa + ((size_t)b * STOT + q0) * HID + h * DH;
  const __bf16* qbBase = Qb + ((size_t)b * STOT + q0) * HID + h * DH;
  const __bf16* kBase  = Kk + (size_t)b * STOT * HID + h * DH;       // K[s][d]
  const __bf16* vBase  = Vt + ((size_t)b * HID + h * DH) * STOT;     // Vt[d][s]

  // preload query A-fragments (two K=32 halves each)
  const v16bf fa0 = fragA_ld(qaBase,      HID, lane);
  const v16bf fa1 = fragA_ld(qaBase + 32, HID, lane);
  const v16bf fb0 = fragA_ld(qbBase,      HID, lane);
  const v16bf fb1 = fragA_ld(qbBase + 32, HID, lane);

  const v8f vz = {0.f,0.f,0.f,0.f,0.f,0.f,0.f,0.f};
  v8f o[4];
#pragma unroll
  for (int nt = 0; nt < 4; ++nt) o[nt] = vz;
  float m[8], l[8];
#pragma unroll
  for (int r = 0; r < 8; ++r) { m[r] = -1e30f; l[r] = 0.f; }

  kv_tile_cp(kBase, vBase, 0, Ks[0], Vs[0], tid);
  cp_wait();
  __syncthreads();

  const int nit = STOT / 32;
  for (int it = 0; it < nit; ++it) {
    const int s0  = it * 32;
    const int buf = it & 1;
    if (it + 1 < nit)
      kv_tile_cp(kBase, vBase, s0 + 32, Ks[buf ^ 1], Vs[buf ^ 1], tid);

    const bool isW = (s0 < SW);                 // key block uniformly word/ent
    const v16bf A0 = isW ? fa0 : fb0;
    const v16bf A1 = isW ? fa1 : fb1;

    v8f sc[2];
#pragma unroll
    for (int g = 0; g < 2; ++g) {
      v16bf B0 = fragB_ld(&Ks[buf][g * 16][0],      64, lane);
      v16bf B1 = fragB_ld(&Ks[buf][g * 16][0] + 32, 64, lane);
      v8f c = vz;
      c = wmma_bf16(A0, B0, c);
      c = wmma_bf16(A1, B1, c);
      sc[g] = c;
    }

    // online softmax (rows r + 8*(lane>>4); reduce across 16-lane halves)
    float corr[8];
#pragma unroll
    for (int r = 0; r < 8; ++r) {
      float v0 = sc[0][r] * 0.125f;             // 1/sqrt(DH)
      float v1 = sc[1][r] * 0.125f;
      float mx = fmaxf(v0, v1);
      mx = fmaxf(mx, __shfl_xor(mx, 1, 32));
      mx = fmaxf(mx, __shfl_xor(mx, 2, 32));
      mx = fmaxf(mx, __shfl_xor(mx, 4, 32));
      mx = fmaxf(mx, __shfl_xor(mx, 8, 32));
      float mn = fmaxf(m[r], mx);
      corr[r]  = __expf(m[r] - mn);
      m[r]     = mn;
      float p0 = __expf(v0 - mn);
      float p1 = __expf(v1 - mn);
      sc[0][r] = p0; sc[1][r] = p1;
      float rs = p0 + p1;
      rs += __shfl_xor(rs, 1, 32);
      rs += __shfl_xor(rs, 2, 32);
      rs += __shfl_xor(rs, 4, 32);
      rs += __shfl_xor(rs, 8, 32);
      l[r] = l[r] * corr[r] + rs;
    }
#pragma unroll
    for (int nt = 0; nt < 4; ++nt)
#pragma unroll
      for (int r = 0; r < 8; ++r) o[nt][r] *= corr[r];

    // stage P (C-layout) into LDS, reload in A-layout (intra-wave only)
#pragma unroll
    for (int g = 0; g < 2; ++g)
#pragma unroll
      for (int r = 0; r < 8; ++r)
        Plds[w][r + ((lane >> 4) << 3)][g * 16 + (lane & 15)] = f2bf(sc[g][r]);
    asm volatile("s_wait_dscnt 0x0" ::: "memory");
    v16bf Pf = fragA_ld(&Plds[w][0][0], 32, lane);

#pragma unroll
    for (int nt = 0; nt < 4; ++nt) {
      v16bf Bv = fragB_ld(&Vs[buf][nt * 16][0], 32, lane);
      o[nt] = wmma_bf16(Pf, Bv, o[nt]);
    }

    cp_wait();
    __syncthreads();
  }

  // normalize + scatter to word/entity output regions (fp32)
#pragma unroll
  for (int r = 0; r < 8; ++r) {
    float inv = 1.0f / l[r];
    int qg = q0 + r + ((lane >> 4) << 3);
#pragma unroll
    for (int nt = 0; nt < 4; ++nt) {
      int dcol = h * DH + nt * 16 + (lane & 15);
      float val = o[nt][r] * inv;
      if (q0 < SW) outW[((size_t)b * SW + qg) * HID + dcol] = val;
      else         outE[((size_t)b * SE + (qg - SW)) * HID + dcol] = val;
    }
  }
}

// ------------------------------ launcher -----------------------------------
extern "C" void kernel_launch(void* const* d_in, const int* in_sizes, int n_in,
                              void* d_out, int out_size, void* d_ws, size_t ws_size,
                              hipStream_t stream) {
  (void)in_sizes; (void)n_in; (void)out_size; (void)ws_size;

  const float* word = (const float*)d_in[0];
  const float* ent  = (const float*)d_in[1];
  // dict order: Wq,bq, Wk,bk, Wv,bv, Ww2e,bw2e, We2w,be2w, We2e,be2e
  const float* Wsrc[6] = { (const float*)d_in[2],  (const float*)d_in[4],
                           (const float*)d_in[6],  (const float*)d_in[8],
                           (const float*)d_in[10], (const float*)d_in[12] };
  const float* bq   = (const float*)d_in[3];
  const float* bk   = (const float*)d_in[5];
  const float* bv   = (const float*)d_in[7];
  const float* bw2e = (const float*)d_in[9];
  const float* be2w = (const float*)d_in[11];
  const float* be2e = (const float*)d_in[13];

  char* ws = (char*)d_ws;
  size_t off = 0;
  auto alloc = [&](size_t bytes) {
    void* p = ws + off;
    off = (off + bytes + 255) & ~(size_t)255;
    return p;
  };
  const size_t actB = (size_t)NBAT * STOT * HID * sizeof(__bf16);
  __bf16* hsB  = (__bf16*)alloc(actB);
  __bf16* Wt[6];
  for (int i = 0; i < 6; ++i) Wt[i] = (__bf16*)alloc((size_t)HID * HID * sizeof(__bf16));
  __bf16* Qa = (__bf16*)alloc(actB);
  __bf16* Qb = (__bf16*)alloc(actB);
  __bf16* Kf = (__bf16*)alloc(actB);
  __bf16* Vt = (__bf16*)alloc(actB);   // [b][d(1024)][s(2304)]

  // 1) convert activations (word+entity -> concat hs) and weights (transposed)
  {
    int totW = NBAT * SW * HID;
    cvt_act<<<(totW + 255) / 256, 256, 0, stream>>>(
        word, hsB, SW * HID, (long)SW * HID, (long)STOT * HID, totW);
    int totE = NBAT * SE * HID;
    cvt_act<<<(totE + 255) / 256, 256, 0, stream>>>(
        ent, hsB + (size_t)SW * HID, SE * HID, (long)SE * HID, (long)STOT * HID, totE);
    for (int i = 0; i < 6; ++i)
      cvt_w_t<<<(HID * HID) / 256, 256, 0, stream>>>(Wsrc[i], Wt[i]);
  }

  // 2) projections (Wt index: 0=Wq 1=Wk 2=Wv 3=Ww2e 4=We2w 5=We2e)
  for (int b = 0; b < NBAT; ++b) {
    const __bf16* hs_b = hsB + (size_t)b * STOT * HID;
    __bf16* qa_b = Qa + (size_t)b * STOT * HID;
    __bf16* qb_b = Qb + (size_t)b * STOT * HID;
    __bf16* k_b  = Kf + (size_t)b * STOT * HID;
    __bf16* vt_b = Vt + (size_t)b * HID * STOT;

    gemm_bf16<<<dim3(8, SW / 128),   256, 0, stream>>>(hs_b, Wt[0], bq,   qa_b, HID, 0);
    gemm_bf16<<<dim3(8, SW / 128),   256, 0, stream>>>(hs_b, Wt[3], bw2e, qb_b, HID, 0);
    gemm_bf16<<<dim3(8, SE / 128),   256, 0, stream>>>(hs_b + (size_t)SW * HID, Wt[4], be2w,
                                                       qa_b + (size_t)SW * HID, HID, 0);
    gemm_bf16<<<dim3(8, SE / 128),   256, 0, stream>>>(hs_b + (size_t)SW * HID, Wt[5], be2e,
                                                       qb_b + (size_t)SW * HID, HID, 0);
    gemm_bf16<<<dim3(8, STOT / 128), 256, 0, stream>>>(hs_b, Wt[1], bk, k_b, HID, 0);
    gemm_bf16<<<dim3(8, STOT / 128), 256, 0, stream>>>(hs_b, Wt[2], bv, vt_b, STOT, 1);
  }

  // 3) attention
  float* outWp = (float*)d_out;
  float* outEp = outWp + (size_t)NBAT * SW * HID;
  attn_fwd<<<NBAT * NHEAD * (STOT / 64), 128, 0, stream>>>(Qa, Qb, Kf, Vt, outWp, outEp);
}
